// SwinModel_4432406249626
// MI455X (gfx1250) — matrix operand order repaired and checked by reading, not compiled, not run on hardware
//
#include <hip/hip_runtime.h>
#include <hip/hip_bf16.h>

// ---------------------------------------------------------------------------
// Swin-B forward for MI455X (gfx1250).
// - All GEMMs + both attention matmuls: v_wmma_f32_16x16x32_f16 (wave32)
// - GEMM tiles: double-buffered GLOBAL_LOAD_ASYNC_TO_LDS_B128 (ASYNCcnt DMA),
//   next tile in flight while current tile is consumed by WMMA
// - Attention Q/K staged with per-lane async DMA (GV mode); V transposed in LDS
// - Activations between GEMMs in f16; residual stream in f32
// ---------------------------------------------------------------------------

typedef __attribute__((ext_vector_type(16))) _Float16 v16h;
typedef __attribute__((ext_vector_type(8)))  _Float16 v8h;
typedef __attribute__((ext_vector_type(8)))  float    v8f;

// 16B async DMA, GVS mode: global (sgpr base + vgpr off) -> LDS[ldsOff]
__device__ __forceinline__ void async_cp16(unsigned ldsOff, unsigned gOff,
                                           unsigned long long base)
{
    asm volatile("global_load_async_to_lds_b128 %0, %1, %2"
                 :: "v"(ldsOff), "v"(gOff), "s"(base) : "memory");
}
// 16B async DMA, GV mode: per-lane 64-bit address
__device__ __forceinline__ void async_cp16_gv(unsigned ldsOff, const void* gaddr)
{
    asm volatile("global_load_async_to_lds_b128 %0, %1, off"
                 :: "v"(ldsOff), "v"((unsigned long long)(uintptr_t)gaddr)
                 : "memory");
}
__device__ __forceinline__ void async_wait0()
{
    asm volatile("s_wait_asynccnt 0x0" ::: "memory");
}
__device__ __forceinline__ void async_wait6()   // tolerate 6 newest in flight
{
    asm volatile("s_wait_asynccnt 0x6" ::: "memory");
}

#define TM 128
#define TN 64
#define TK 32

// ---------------------------------------------------------------------------
// GEMM: out[M,N] = act(A[M,K] @ W[N,K]^T + bias) (+ residual)
// f16 inputs, f32 accumulate. 128 threads = 4 waves; wave computes 32x64.
// Double-buffered async-to-LDS pipeline.
// ---------------------------------------------------------------------------
template <int ACT, int HAS_BIAS, int HAS_RES, int OUT16>
__global__ __launch_bounds__(128) void gemm_wmma(
    const _Float16* __restrict__ A, const _Float16* __restrict__ W,
    const float* __restrict__ bias, const float* __restrict__ resid,
    float* out32, _Float16* out16, int M, int N, int K)
{
    __shared__ _Float16 As[2][TM][TK];
    __shared__ _Float16 Bs[2][TN][TK];
    const unsigned aStride = TM * TK * 2;   // bytes per A buffer
    const unsigned bStride = TN * TK * 2;   // bytes per B buffer

    const int tid  = threadIdx.x;
    const int wave = tid >> 5;
    const int lane = tid & 31;
    const int hi   = lane >> 4;
    const int lm   = lane & 15;
    const int gm0  = blockIdx.y * TM;
    const int gn0  = blockIdx.x * TN;

    const unsigned long long baseA = (unsigned long long)(uintptr_t)A;
    const unsigned long long baseW = (unsigned long long)(uintptr_t)W;

    // A staging: thread -> tile row tid, 4 chunks of 8 halves (16B).
    // OOB rows clamped to M-1 (addresses stay in-buffer, results discarded).
    const int arow = (gm0 + tid < M) ? (gm0 + tid) : (M - 1);
    const unsigned aOffBase = (unsigned)((size_t)arow * K * 2);
    unsigned ldsA[4];
#pragma unroll
    for (int c = 0; c < 4; ++c)
        ldsA[c] = (unsigned)(uintptr_t)&As[0][tid][c * 8];

    // B staging: 2 chunks/thread; idx = tid*2+c -> row idx>>2, chunk idx&3.
    unsigned ldsB[2], bOffBase[2]; int bch[2];
#pragma unroll
    for (int c = 0; c < 2; ++c) {
        const int idx  = tid * 2 + c;
        const int brow = idx >> 2;
        bch[c] = idx & 3;
        bOffBase[c] = (unsigned)((size_t)(gn0 + brow) * K * 2);
        ldsB[c] = (unsigned)(uintptr_t)&Bs[0][brow][bch[c] * 8];
    }

    v8f acc[8] = {};   // [sub(2)][nt(4)]
    const int nTiles = K / TK;

    // preload tile 0 into buffer 0
#pragma unroll
    for (int c = 0; c < 4; ++c)
        async_cp16(ldsA[c], aOffBase + (unsigned)(c * 16), baseA);
#pragma unroll
    for (int c = 0; c < 2; ++c)
        async_cp16(ldsB[c], bOffBase[c] + (unsigned)(bch[c] * 16), baseW);

    for (int i = 0; i < nTiles; ++i) {
        const int cur = i & 1;
        const unsigned nb = (unsigned)(cur ^ 1);
        if (i + 1 < nTiles) {               // issue tile i+1 into other buffer
            const unsigned k1 = (unsigned)((i + 1) * TK * 2);
#pragma unroll
            for (int c = 0; c < 4; ++c)
                async_cp16(ldsA[c] + nb * aStride,
                           aOffBase + k1 + (unsigned)(c * 16), baseA);
#pragma unroll
            for (int c = 0; c < 2; ++c)
                async_cp16(ldsB[c] + nb * bStride,
                           bOffBase[c] + k1 + (unsigned)(bch[c] * 16), baseW);
            async_wait6();                  // tile i resident; i+1 in flight
        } else {
            async_wait0();
        }
        __syncthreads();

        const _Float16 (*Ac)[TK] = As[cur];
        const _Float16 (*Bc)[TK] = Bs[cur];

        v16h a0, a1;
        const int am0 = wave * 32 + lm;
#pragma unroll
        for (int e = 0; e < 16; ++e) {
            const int kk = e + (hi ? 16 : 0);
            a0[e] = Ac[am0][kk];
            a1[e] = Ac[am0 + 16][kk];
        }
#pragma unroll
        for (int nt = 0; nt < 4; ++nt) {
            v16h b;
            const int bn = nt * 16 + lm;
#pragma unroll
            for (int e = 0; e < 16; ++e)
                b[e] = Bc[bn][e + (hi ? 16 : 0)];
            acc[nt]     = __builtin_amdgcn_wmma_f32_16x16x32_f16(
                false, a0, false, b, (short)0, acc[nt], false, false);
            acc[4 + nt] = __builtin_amdgcn_wmma_f32_16x16x32_f16(
                false, a1, false, b, (short)0, acc[4 + nt], false, false);
        }
        __syncthreads();   // all waves done with buf[cur] before it is refilled
    }

    // epilogue (all mode checks are compile-time)
#pragma unroll
    for (int sub = 0; sub < 2; ++sub) {
#pragma unroll
        for (int nt = 0; nt < 4; ++nt) {
            const int col = gn0 + nt * 16 + lm;           // N % 64 == 0 always
            const float bv = HAS_BIAS ? bias[col] : 0.f;
            const int rbase = gm0 + wave * 32 + sub * 16 + (hi ? 8 : 0);
#pragma unroll
            for (int r = 0; r < 8; ++r) {
                const int row = rbase + r;
                if (row < M) {
                    float v = acc[sub * 4 + nt][r] + bv;
                    if (ACT) v = 0.5f * v * (1.f + erff(v * 0.70710678118654752f));
                    if (HAS_RES) v += resid[(size_t)row * N + col];
                    if (OUT16) out16[(size_t)row * N + col] = (_Float16)v;
                    else       out32[(size_t)row * N + col] = v;
                }
            }
        }
    }
}

// ---------------------------------------------------------------------------
// Windowed attention, one wave per (window, head). head_dim == 32.
// Q/K async-DMA'd to LDS (per-lane GV mode); V transposed into LDS so the
// P@V B-fragments are contiguous. q-scale folded into softmax.
// ---------------------------------------------------------------------------
__global__ __launch_bounds__(32) void attn_win(
    const _Float16* __restrict__ qkv, _Float16* o, const float* __restrict__ rpb,
    int Hd, int Wd, int C, int heads, int s)
{
    __shared__ v8h Qs8[64][4];
    __shared__ v8h Ks8[64][4];
    __shared__ _Float16 Vt[32][64];        // [head_dim][token]
    __shared__ float    Sf[64][64];
    __shared__ _Float16 Ps[64][64];
    _Float16* Qs = (_Float16*)Qs8;
    _Float16* Ks = (_Float16*)Ks8;

    const int win  = blockIdx.x;
    const int head = blockIdx.y;
    const int nW = Wd / 7, nH = Hd / 7;
    const int b  = win / (nH * nW);
    const int wr = win % (nH * nW);
    const int wy = wr / nW, wx = wr % nW;
    const int lane = threadIdx.x;
    const int hi = lane >> 4, lm = lane & 15;
    const float scale = 0.17677669529663687f;   // 32^-0.5

    for (int t = lane; t < 64; t += 32) {
        if (t < 49) {
            const int ty = t / 7, tx = t % 7;
            const int hh = (wy * 7 + ty + s) % Hd;
            const int ww = (wx * 7 + tx + s) % Wd;
            const size_t tok = ((size_t)b * Hd + hh) * Wd + ww;
            const _Float16* base = qkv + tok * (size_t)(3 * C) + head * 32;
#pragma unroll
            for (int j = 0; j < 4; ++j) {
                async_cp16_gv((unsigned)(uintptr_t)&Qs8[t][j], base + j * 8);
                async_cp16_gv((unsigned)(uintptr_t)&Ks8[t][j], base + C + j * 8);
            }
            const v8h* v8p = (const v8h*)(base + 2 * C);
#pragma unroll
            for (int j = 0; j < 4; ++j) {
                const v8h vv = v8p[j];
#pragma unroll
                for (int d = 0; d < 8; ++d) Vt[j * 8 + d][t] = vv[d];
            }
        } else {
            const v8h z = {};
#pragma unroll
            for (int j = 0; j < 4; ++j) { Qs8[t][j] = z; Ks8[t][j] = z; }
#pragma unroll
            for (int d = 0; d < 32; ++d) Vt[d][t] = (_Float16)0.f;
        }
    }
    async_wait0();
    __syncthreads();

    // S = Q @ K^T : 4x4 tiles, K=32 each
#pragma unroll
    for (int mt = 0; mt < 4; ++mt) {
        v16h a;
#pragma unroll
        for (int e = 0; e < 16; ++e) {
            const int kk = e + (hi ? 16 : 0);
            a[e] = Qs[(mt * 16 + lm) * 32 + kk];
        }
#pragma unroll
        for (int nt = 0; nt < 4; ++nt) {
            v16h bf;
#pragma unroll
            for (int e = 0; e < 16; ++e)
                bf[e] = Ks[(nt * 16 + lm) * 32 + e + (hi ? 16 : 0)];
            v8f c = {};
            c = __builtin_amdgcn_wmma_f32_16x16x32_f16(
                false, a, false, bf, (short)0, c, false, false);
#pragma unroll
            for (int r = 0; r < 8; ++r)
                Sf[mt * 16 + r + (hi ? 8 : 0)][nt * 16 + lm] = c[r];
        }
    }
    __syncthreads();

    // softmax rows: scale + rel-pos bias + shifted-window mask
    for (int row = lane; row < 64; row += 32) {
        if (row < 49) {
            const int yi = row / 7, xi = row % 7;
            int ri = 0;
            if (s) {
                const int gh = wy * 7 + yi, gw = wx * 7 + xi;
                const int rh = (gh < Hd - 7) ? 0 : ((gh < Hd - 3) ? 1 : 2);
                const int rw = (gw < Wd - 7) ? 0 : ((gw < Wd - 3) ? 1 : 2);
                ri = rh * 3 + rw;
            }
            float mx = -1e30f;
            for (int col = 0; col < 49; ++col) {
                const int yj = col / 7, xj = col % 7;
                const int idx = (yi - yj + 6) * 13 + (xi - xj + 6);
                float v = Sf[row][col] * scale + rpb[idx * heads + head];
                if (s) {
                    const int gh = wy * 7 + yj, gw = wx * 7 + xj;
                    const int rh = (gh < Hd - 7) ? 0 : ((gh < Hd - 3) ? 1 : 2);
                    const int rw = (gw < Wd - 7) ? 0 : ((gw < Wd - 3) ? 1 : 2);
                    if (rh * 3 + rw != ri) v -= 100.f;
                }
                Sf[row][col] = v;
                mx = fmaxf(mx, v);
            }
            float sum = 0.f;
            for (int col = 0; col < 49; ++col) sum += __expf(Sf[row][col] - mx);
            const float inv = 1.f / sum;
            for (int col = 0; col < 49; ++col)
                Ps[row][col] = (_Float16)(__expf(Sf[row][col] - mx) * inv);
            for (int col = 49; col < 64; ++col) Ps[row][col] = (_Float16)0.f;
        } else {
            for (int col = 0; col < 64; ++col) Ps[row][col] = (_Float16)0.f;
        }
    }
    __syncthreads();

    // O = P @ V : 4 m-tiles x 2 n-tiles, K=64 (two WMMA k-steps)
#pragma unroll
    for (int mt = 0; mt < 4; ++mt) {
#pragma unroll
        for (int nt = 0; nt < 2; ++nt) {
            v8f c = {};
#pragma unroll
            for (int kt = 0; kt < 2; ++kt) {
                v16h a, bf;
#pragma unroll
                for (int e = 0; e < 16; ++e) {
                    const int kk = e + (hi ? 16 : 0);
                    a[e]  = Ps[mt * 16 + lm][kt * 32 + kk];
                    bf[e] = Vt[nt * 16 + lm][kt * 32 + e + (hi ? 16 : 0)];
                }
                c = __builtin_amdgcn_wmma_f32_16x16x32_f16(
                    false, a, false, bf, (short)0, c, false, false);
            }
#pragma unroll
            for (int r = 0; r < 8; ++r) {
                const int t = mt * 16 + r + (hi ? 8 : 0);
                if (t < 49) {
                    const int ty = t / 7, tx = t % 7;
                    const int hh = (wy * 7 + ty + s) % Hd;
                    const int ww = (wx * 7 + tx + s) % Wd;
                    const size_t tok = ((size_t)b * Hd + hh) * Wd + ww;
                    o[tok * (size_t)C + head * 32 + nt * 16 + lm] = (_Float16)c[r];
                }
            }
        }
    }
}

// ---------------------------------------------------------------------------
// Small kernels (negligible FLOPs)
// ---------------------------------------------------------------------------
__global__ void cvt_f16(const float* __restrict__ src, _Float16* dst, size_t n)
{
    const size_t i = (size_t)blockIdx.x * blockDim.x + threadIdx.x;
    if (i < n) dst[i] = (_Float16)src[i];
}

__global__ void patch_embed(const float* __restrict__ img,
                            const float* __restrict__ w,
                            const float* __restrict__ bias, float* out)
{
    size_t i = (size_t)blockIdx.x * blockDim.x + threadIdx.x;
    const size_t total = (size_t)32 * 56 * 56 * 128;
    if (i >= total) return;
    const int oc = (int)(i & 127);
    size_t t = i >> 7;
    const int ow = (int)(t % 56); t /= 56;
    const int oh = (int)(t % 56); t /= 56;
    const int b  = (int)t;
    float acc = bias[oc];
#pragma unroll
    for (int ci = 0; ci < 3; ++ci)
#pragma unroll
        for (int kh = 0; kh < 4; ++kh)
#pragma unroll
            for (int kw = 0; kw < 4; ++kw) {
                const float xv = img[(((size_t)b * 3 + ci) * 224 + (oh * 4 + kh)) * 224
                                     + (ow * 4 + kw)];
                const float wv = w[((oc * 3 + ci) * 4 + kh) * 4 + kw];
                acc += xv * wv;
            }
    out[i] = acc;
}

// LayerNorm: one wave32 per row; OUT16 selects f16 vs f32 output
template <int OUT16>
__global__ __launch_bounds__(32) void layernorm_k(
    const float* __restrict__ x, const float* __restrict__ g,
    const float* __restrict__ bt, float* out32, _Float16* out16, int C)
{
    const int row  = blockIdx.x;
    const int lane = threadIdx.x;
    const float* xr = x + (size_t)row * C;
    float s = 0.f;
    for (int c = lane; c < C; c += 32) s += xr[c];
#pragma unroll
    for (int off = 16; off; off >>= 1) s += __shfl_xor(s, off, 32);
    const float m = s / (float)C;
    float v = 0.f;
    for (int c = lane; c < C; c += 32) { const float d = xr[c] - m; v += d * d; }
#pragma unroll
    for (int off = 16; off; off >>= 1) v += __shfl_xor(v, off, 32);
    const float r = rsqrtf(v / (float)C + 1e-5f);
    for (int c = lane; c < C; c += 32) {
        const float y = (xr[c] - m) * r * g[c] + bt[c];
        if (OUT16) out16[(size_t)row * C + c] = (_Float16)y;
        else       out32[(size_t)row * C + c] = y;
    }
}

__global__ void merge_gather(const float* __restrict__ x, float* out,
                             int B, int H, int W, int C)
{
    size_t i = (size_t)blockIdx.x * blockDim.x + threadIdx.x;
    const int C4 = 4 * C;
    const size_t total = (size_t)B * (H / 2) * (W / 2) * C4;
    if (i >= total) return;
    const int c4 = (int)(i % C4); size_t t = i / C4;
    const int j  = (int)(t % (W / 2)); t /= (W / 2);
    const int ii = (int)(t % (H / 2)); t /= (H / 2);
    const int b  = (int)t;
    const int which = c4 / C, c = c4 % C;
    const int dh = which & 1;
    const int dw = (which >> 1) & 1;
    out[i] = x[(((size_t)b * H + (2 * ii + dh)) * W + (2 * j + dw)) * C + c];
}

__global__ void mean_pool(const float* __restrict__ h, float* out)
{
    const int i = blockIdx.x * blockDim.x + threadIdx.x;
    if (i >= 32 * 1024) return;
    const int c = i & 1023, b = i >> 10;
    float s = 0.f;
    for (int t = 0; t < 49; ++t) s += h[((size_t)b * 49 + t) * 1024 + c];
    out[i] = s * (1.f / 49.f);
}

// ---------------------------------------------------------------------------
// Host orchestration
// ---------------------------------------------------------------------------
extern "C" void kernel_launch(void* const* d_in, const int* in_sizes, int n_in,
                              void* d_out, int out_size, void* d_ws, size_t ws_size,
                              hipStream_t stream)
{
    (void)in_sizes; (void)n_in; (void)out_size; (void)ws_size;

    const float* img     = (const float*)d_in[0];
    const float* patch_w = (const float*)d_in[1];
    const float* patch_b = (const float*)d_in[2];
    const float* pn_g    = (const float*)d_in[3];
    const float* pn_b    = (const float*)d_in[4];
    const float* fn_g    = (const float*)d_in[5];
    const float* fn_b    = (const float*)d_in[6];

    // workspace layout
    char* ws = (char*)d_ws;
    const size_t ELX = (size_t)100352 * 128;
    float*    bx   = (float*)ws;                          // residual (f32)
    float*    bt   = (float*)(ws + ELX * 4);              // f32 temp
    _Float16* b16a = (_Float16*)(ws + ELX * 8);           // f16, M*C max
    _Float16* b16b = (_Float16*)(ws + ELX * 8 + ELX * 2); // f16, M*4C max
    _Float16* wscr = (_Float16*)(ws + ELX * 8 + ELX * 2 + ELX * 8); // weights

    auto cvt = [&](const float* src, size_t n) -> _Float16* {
        cvt_f16<<<(unsigned)((n + 255) / 256), 256, 0, stream>>>(src, wscr, n);
        return wscr;
    };

    const int B = 32;
    const int DEPTH[4] = {2, 2, 18, 2};
    const int HEADS[4] = {4, 8, 16, 32};

    {
        const size_t total = (size_t)B * 56 * 56 * 128;
        patch_embed<<<(unsigned)((total + 255) / 256), 256, 0, stream>>>(
            img, patch_w, patch_b, bt);
        layernorm_k<0><<<B * 56 * 56, 32, 0, stream>>>(bt, pn_g, pn_b, bx, nullptr, 128);
    }

    int H = 56, C = 128;
    int p  = 7;
    int pm = 7 + 13 * 24;
    for (int si = 0; si < 4; ++si) {
        const int heads = HEADS[si];
        const int M = B * H * H;
        for (int bi = 0; bi < DEPTH[si]; ++bi) {
            const float* n1g    = (const float*)d_in[p + 0];
            const float* n1b    = (const float*)d_in[p + 1];
            const float* qkv_w  = (const float*)d_in[p + 2];
            const float* qkv_b  = (const float*)d_in[p + 3];
            const float* rpb    = (const float*)d_in[p + 4];
            const float* proj_w = (const float*)d_in[p + 5];
            const float* proj_b = (const float*)d_in[p + 6];
            const float* n2g    = (const float*)d_in[p + 7];
            const float* n2b    = (const float*)d_in[p + 8];
            const float* fc1_w  = (const float*)d_in[p + 9];
            const float* fc1_b  = (const float*)d_in[p + 10];
            const float* fc2_w  = (const float*)d_in[p + 11];
            const float* fc2_b  = (const float*)d_in[p + 12];
            p += 13;

            const int s = (H > 7 && (bi & 1)) ? 3 : 0;

            layernorm_k<1><<<M, 32, 0, stream>>>(bx, n1g, n1b, nullptr, b16a, C);
            {   // qkv: bias, no res, out f16
                _Float16* wq = cvt(qkv_w, (size_t)3 * C * C);
                dim3 g((3 * C) / TN, (M + TM - 1) / TM);
                gemm_wmma<0, 1, 0, 1><<<g, 128, 0, stream>>>(
                    b16a, wq, qkv_b, nullptr, nullptr, b16b, M, 3 * C, C);
            }
            {   // windowed attention
                dim3 g((unsigned)(B * (H / 7) * (H / 7)), (unsigned)heads);
                attn_win<<<g, 32, 0, stream>>>(b16b, b16a, rpb, H, H, C, heads, s);
            }
            {   // proj: bias, residual, out f32
                _Float16* wp = cvt(proj_w, (size_t)C * C);
                dim3 g(C / TN, (M + TM - 1) / TM);
                gemm_wmma<0, 1, 1, 0><<<g, 128, 0, stream>>>(
                    b16a, wp, proj_b, bx, bx, nullptr, M, C, C);
            }
            layernorm_k<1><<<M, 32, 0, stream>>>(bx, n2g, n2b, nullptr, b16a, C);
            {   // fc1: bias, GELU, out f16
                _Float16* w1 = cvt(fc1_w, (size_t)4 * C * C);
                dim3 g((4 * C) / TN, (M + TM - 1) / TM);
                gemm_wmma<1, 1, 0, 1><<<g, 128, 0, stream>>>(
                    b16a, w1, fc1_b, nullptr, nullptr, b16b, M, 4 * C, C);
            }
            {   // fc2: bias, residual, out f32
                _Float16* w2 = cvt(fc2_w, (size_t)4 * C * C);
                dim3 g(C / TN, (M + TM - 1) / TM);
                gemm_wmma<0, 1, 1, 0><<<g, 128, 0, stream>>>(
                    b16b, w2, fc2_b, bx, bx, nullptr, M, C, 4 * C);
            }
        }
        if (si < 3) {
            const float* mg    = (const float*)d_in[pm + 0];
            const float* mb    = (const float*)d_in[pm + 1];
            const float* red_w = (const float*)d_in[pm + 2];
            pm += 3;
            const int M2 = B * (H / 2) * (H / 2);
            const size_t total = (size_t)M2 * 4 * C;
            merge_gather<<<(unsigned)((total + 255) / 256), 256, 0, stream>>>(
                bx, bt, B, H, H, C);
            layernorm_k<1><<<M2, 32, 0, stream>>>(bt, mg, mb, nullptr, b16b, 4 * C);
            _Float16* wr = cvt(red_w, (size_t)8 * C * C);
            dim3 g((2 * C) / TN, (M2 + TM - 1) / TM);
            gemm_wmma<0, 0, 0, 0><<<g, 128, 0, stream>>>(
                b16b, wr, nullptr, nullptr, bx, nullptr, M2, 2 * C, 4 * C);
            H /= 2; C *= 2;
        }
    }

    layernorm_k<0><<<B * 49, 32, 0, stream>>>(bx, fn_g, fn_b, bt, nullptr, 1024);
    mean_pool<<<(32 * 1024 + 255) / 256, 256, 0, stream>>>(bt, (float*)d_out);
}
